// MultiHeadSelfAttention_75325136437652
// MI455X (gfx1250) — matrix operand order repaired and checked by reading, not compiled, not run on hardware
//
#include <hip/hip_runtime.h>

typedef __bf16 bf16;
typedef __attribute__((ext_vector_type(16))) __bf16 v16bf;
typedef __attribute__((ext_vector_type(8)))  __bf16 bf16x8;
typedef __attribute__((ext_vector_type(4)))  __bf16 bf16x4;
typedef __attribute__((ext_vector_type(8)))  float   v8f;

constexpr int BATCH = 4, S = 2048, Dm = 1024, H = 16, DK = 64;

#if defined(__gfx1250__) &&                                              \
    __has_builtin(__builtin_amdgcn_global_load_async_to_lds_b128) &&     \
    __has_builtin(__builtin_amdgcn_s_wait_asynccnt)
#define HAVE_ASYNC_LDS 1
#else
#define HAVE_ASYNC_LDS 0
#endif

#if HAVE_ASYNC_LDS
typedef int v4i_gcc __attribute__((vector_size(16)));
typedef __attribute__((address_space(1))) v4i_gcc* as1_v4i;
typedef __attribute__((address_space(3))) v4i_gcc* as3_v4i;
#endif

__device__ __forceinline__ v8f zero8() {
  v8f z = {0.f, 0.f, 0.f, 0.f, 0.f, 0.f, 0.f, 0.f};
  return z;
}

// Copy 16 bytes global -> LDS. Async (ASYNCcnt-tracked) when available.
__device__ __forceinline__ void cp_b128(bf16* lds_dst, const bf16* gsrc) {
#if HAVE_ASYNC_LDS
  __builtin_amdgcn_global_load_async_to_lds_b128((as1_v4i)gsrc, (as3_v4i)lds_dst,
                                                 0, 0);
#else
  *(bf16x8*)lds_dst = *(const bf16x8*)gsrc;
#endif
}

__device__ __forceinline__ void wait_async_copies() {
#if HAVE_ASYNC_LDS
  __builtin_amdgcn_s_wait_asynccnt(0);
#endif
}

// ---- Fragment loaders matching CDNA5 16-bit WMMA VGPR layouts (wave32) ----
__device__ __forceinline__ v16bf load_a_frag(const bf16* src, int ld) {
  const int lane = threadIdx.x & 31;
  const int m  = lane & 15;
  const int k0 = (lane & 16) >> 1;               // 0 or 8
  union { v16bf v; bf16x8 h[2]; } u;
  u.h[0] = *(const bf16x8*)(src + (size_t)m * ld + k0);
  u.h[1] = *(const bf16x8*)(src + (size_t)m * ld + k0 + 16);
  return u.v;
}

// B (32x16, KxN) whose SOURCE is stored row-major [N][K] (i.e. B^T in memory).
__device__ __forceinline__ v16bf load_bt_frag(const bf16* src, int ld) {
  const int lane = threadIdx.x & 31;
  const int n  = lane & 15;
  const int k0 = (lane & 16) >> 1;               // 0 or 8
  union { v16bf v; bf16x8 h[2]; } u;
  u.h[0] = *(const bf16x8*)(src + (size_t)n * ld + k0);
  u.h[1] = *(const bf16x8*)(src + (size_t)n * ld + k0 + 16);
  return u.v;
}

__device__ __forceinline__ v8f wmma_bf16(v16bf a, v16bf b, v8f c) {
  return __builtin_amdgcn_wmma_f32_16x16x32_bf16(false, a, false, b, (short)0, c,
                                                 false, false);
}

// ---------------- Kernel 1: fp32 -> bf16 cast (x and weights) ----------------
__global__ void cvt_f32_bf16_v4(const float* __restrict__ src,
                                bf16* __restrict__ dst, int n4) {
  for (int i = blockIdx.x * blockDim.x + threadIdx.x; i < n4;
       i += gridDim.x * blockDim.x) {
    const float4 f = ((const float4*)src)[i];
    bf16x4 o = {(bf16)f.x, (bf16)f.y, (bf16)f.z, (bf16)f.w};
    ((bf16x4*)dst)[i] = o;
  }
}

// ---------------- Kernel 2: fused QKV projection ----------------
// grid = (M/128, D/64, 3); block = 128 (4 waves, each owns a 32x64 tile).
// Double-buffered LDS: async loads of tile kt+1 overlap WMMA on tile kt.
__global__ void __launch_bounds__(128)
qkv_gemm_kernel(const bf16* __restrict__ xh, const bf16* __restrict__ wh,
                const float* __restrict__ bq, const float* __restrict__ bk,
                const float* __restrict__ bv, bf16* __restrict__ qh,
                bf16* __restrict__ kh, bf16* __restrict__ vt) {
  __shared__ __align__(16) bf16 xs[2][128][40];
  __shared__ __align__(16) bf16 wsm[2][64][40];
  const int tid = threadIdx.x, wid = tid >> 5, lane = tid & 31;
  const int m0 = blockIdx.x * 128;
  const int n0 = blockIdx.y * 64;
  const int z  = blockIdx.z;
  const bf16*  W    = wh + (size_t)z * Dm * Dm;
  const float* bias = (z == 0) ? bq : (z == 1) ? bk : bv;

  auto stage = [&](int buf, int kt) {
    const int k0 = kt * 32;
#pragma unroll
    for (int i = 0; i < 4; ++i) {                // 128x32 x tile, 4 chunks/thd
      int j = tid * 4 + i;
      int r = j >> 2, c = (j & 3) * 8;
      cp_b128(&xs[buf][r][c], xh + (size_t)(m0 + r) * Dm + k0 + c);
    }
#pragma unroll
    for (int i = 0; i < 2; ++i) {                // 64x32 W tile, 2 chunks/thd
      int j = tid * 2 + i;
      int r = j >> 2, c = (j & 3) * 8;
      cp_b128(&wsm[buf][r][c], W + (size_t)(n0 + r) * Dm + k0 + c);
    }
  };

  v8f acc[2][4];
#pragma unroll
  for (int mi = 0; mi < 2; ++mi)
#pragma unroll
    for (int nc = 0; nc < 4; ++nc) acc[mi][nc] = zero8();

  stage(0, 0);
  int cur = 0;
  for (int kt = 0; kt < Dm / 32; ++kt) {
    wait_async_copies();
    __syncthreads();                             // buf[cur] ready everywhere
    if (kt + 1 < Dm / 32) stage(cur ^ 1, kt + 1);

    v16bf a[2];
#pragma unroll
    for (int mi = 0; mi < 2; ++mi)
      a[mi] = load_a_frag(&xs[cur][wid * 32 + mi * 16][0], 40);
#pragma unroll
    for (int nc = 0; nc < 4; ++nc) {
      v16bf bfrag = load_bt_frag(&wsm[cur][nc * 16][0], 40);
#pragma unroll
      for (int mi = 0; mi < 2; ++mi)
        acc[mi][nc] = wmma_bf16(a[mi], bfrag, acc[mi][nc]);
    }
    cur ^= 1;
  }

  const int nl = lane & 15;
  const int mh = (lane & 16) ? 8 : 0;
#pragma unroll
  for (int mi = 0; mi < 2; ++mi)
#pragma unroll
    for (int nc = 0; nc < 4; ++nc)
#pragma unroll
      for (int v = 0; v < 8; ++v) {
        int gm = m0 + wid * 32 + mi * 16 + mh + v;   // row in [0, B*S)
        int n  = n0 + nc * 16 + nl;                  // col in [0, D)
        float val = acc[mi][nc][v] + bias[n];
        int bb = gm >> 11, s = gm & (S - 1);
        int h  = n >> 6,  dk = n & (DK - 1);
        if (z == 0)
          qh[(((size_t)(bb * H + h)) * S + s) * DK + dk] = (bf16)val;
        else if (z == 1)
          kh[(((size_t)(bb * H + h)) * S + s) * DK + dk] = (bf16)val;
        else
          vt[(((size_t)(bb * H + h)) * DK + dk) * S + s] = (bf16)val;
      }
}

// ---------------- Kernel 3: flash attention ----------------
// grid = (S/64, B*H); block = 128 (4 waves, each owns 16 query rows).
// Double-buffered 64-key tiles; async fill overlaps scores+softmax+PV.
__global__ void __launch_bounds__(128)
attn_kernel(const bf16* __restrict__ qh, const bf16* __restrict__ kh,
            const bf16* __restrict__ vt, bf16* __restrict__ ctx) {
  __shared__ __align__(16) bf16 Ks[2][64][72];   // K tile   [t][dk]
  __shared__ __align__(16) bf16 Vts[2][64][72];  // V^T tile [dk][t]
  __shared__ __align__(16) bf16 Ps[4][16][72];   // per-wave P re-layout
  const int tid = threadIdx.x, wid = tid >> 5, lane = tid & 31;
  const int bh = blockIdx.y;
  const int b = bh >> 4, h = bh & 15;
  const int s0 = blockIdx.x * 64 + wid * 16;

  const bf16* Qb = qh + ((size_t)bh * S + s0) * DK;
  v16bf qf[2];
#pragma unroll
  for (int kc = 0; kc < 2; ++kc) qf[kc] = load_a_frag(Qb + kc * 32, DK);

  float mrow[8], lrow[8];
  v8f oacc[4];
#pragma unroll
  for (int v = 0; v < 8; ++v) { mrow[v] = -3.0e38f; lrow[v] = 0.f; }
#pragma unroll
  for (int nc = 0; nc < 4; ++nc) oacc[nc] = zero8();

  const bf16* Kbase = kh + (size_t)bh * S * DK;
  const bf16* Vbase = vt + (size_t)bh * DK * S;

  auto stage = [&](int buf, int t0) {
#pragma unroll
    for (int i = 0; i < 4; ++i) {                // 64x64 each, 4+4 chunks/thd
      int j = tid * 4 + i;
      int r = j >> 3, c = (j & 7) * 8;
      cp_b128(&Ks[buf][r][c],  Kbase + (size_t)(t0 + r) * DK + c);
      cp_b128(&Vts[buf][r][c], Vbase + (size_t)r * S + t0 + c);
    }
  };

  stage(0, 0);
  int cur = 0;
  for (int t0 = 0; t0 < S; t0 += 64) {
    wait_async_copies();
    __syncthreads();                             // buf[cur] ready everywhere
    if (t0 + 64 < S) stage(cur ^ 1, t0 + 64);

    // scores: 4 tiles of 16x16, K-dim = DK = 64 (2 WMMAs each)
    v8f sc[4];
#pragma unroll
    for (int st = 0; st < 4; ++st) {
      sc[st] = zero8();
#pragma unroll
      for (int kc = 0; kc < 2; ++kc) {
        v16bf bfrag = load_bt_frag(&Ks[cur][st * 16][kc * 32], 72);
        sc[st] = wmma_bf16(qf[kc], bfrag, sc[st]);
      }
    }

    // online softmax update (rows live across 16-lane halves)
    float tmax[8];
#pragma unroll
    for (int v = 0; v < 8; ++v) {
      float t = -3.0e38f;
#pragma unroll
      for (int st = 0; st < 4; ++st) {
        sc[st][v] *= 0.125f;                     // 1/sqrt(64)
        t = fmaxf(t, sc[st][v]);
      }
      tmax[v] = t;
    }
#pragma unroll
    for (int off = 1; off <= 8; off <<= 1)
#pragma unroll
      for (int v = 0; v < 8; ++v)
        tmax[v] = fmaxf(tmax[v], __shfl_xor(tmax[v], off, 32));

    float alpha[8], rs[8];
#pragma unroll
    for (int v = 0; v < 8; ++v) {
      float mn = fmaxf(mrow[v], tmax[v]);
      alpha[v] = __expf(mrow[v] - mn);
      mrow[v]  = mn;
      int m = ((lane & 16) ? 8 : 0) + v;
      float acc_rs = 0.f;
#pragma unroll
      for (int st = 0; st < 4; ++st) {
        float p = __expf(sc[st][v] - mn);
        acc_rs += p;
        Ps[wid][m][st * 16 + (lane & 15)] = (bf16)p;
      }
      rs[v] = acc_rs;
    }
#pragma unroll
    for (int off = 1; off <= 8; off <<= 1)
#pragma unroll
      for (int v = 0; v < 8; ++v) rs[v] += __shfl_xor(rs[v], off, 32);
#pragma unroll
    for (int v = 0; v < 8; ++v) lrow[v] = lrow[v] * alpha[v] + rs[v];
#pragma unroll
    for (int nc = 0; nc < 4; ++nc)
#pragma unroll
      for (int v = 0; v < 8; ++v) oacc[nc][v] *= alpha[v];

    __syncthreads();                             // make Ps visible across lanes

    // O += P (16x64) x V (64x64): 8 WMMAs
    v16bf pf[2];
#pragma unroll
    for (int kc = 0; kc < 2; ++kc)
      pf[kc] = load_a_frag(&Ps[wid][0][kc * 32], 72);
#pragma unroll
    for (int nc = 0; nc < 4; ++nc)
#pragma unroll
      for (int kc = 0; kc < 2; ++kc) {
        v16bf bfrag = load_bt_frag(&Vts[cur][nc * 16][kc * 32], 72);
        oacc[nc] = wmma_bf16(pf[kc], bfrag, oacc[nc]);
      }
    cur ^= 1;
  }

  float rinv[8];
#pragma unroll
  for (int v = 0; v < 8; ++v) rinv[v] = 1.0f / lrow[v];
#pragma unroll
  for (int nc = 0; nc < 4; ++nc)
#pragma unroll
    for (int v = 0; v < 8; ++v) {
      int s   = s0 + ((lane & 16) ? 8 : 0) + v;
      int col = h * DK + nc * 16 + (lane & 15);
      ctx[((size_t)b * S + s) * Dm + col] = (bf16)(oacc[nc][v] * rinv[v]);
    }
}

// ---------------- Kernel 4: output projection ----------------
// grid = (M/128, D/64); block = 128. out = ctx @ Wo^T + bo (fp32 out).
__global__ void __launch_bounds__(128)
oproj_kernel(const bf16* __restrict__ ctx, const bf16* __restrict__ Wo,
             const float* __restrict__ bo, float* __restrict__ out) {
  __shared__ __align__(16) bf16 cs[2][128][40];
  __shared__ __align__(16) bf16 wsm[2][64][40];
  const int tid = threadIdx.x, wid = tid >> 5, lane = tid & 31;
  const int m0 = blockIdx.x * 128;
  const int n0 = blockIdx.y * 64;

  auto stage = [&](int buf, int kt) {
    const int k0 = kt * 32;
#pragma unroll
    for (int i = 0; i < 4; ++i) {
      int j = tid * 4 + i;
      int r = j >> 2, c = (j & 3) * 8;
      cp_b128(&cs[buf][r][c], ctx + (size_t)(m0 + r) * Dm + k0 + c);
    }
#pragma unroll
    for (int i = 0; i < 2; ++i) {
      int j = tid * 2 + i;
      int r = j >> 2, c = (j & 3) * 8;
      cp_b128(&wsm[buf][r][c], Wo + (size_t)(n0 + r) * Dm + k0 + c);
    }
  };

  v8f acc[2][4];
#pragma unroll
  for (int mi = 0; mi < 2; ++mi)
#pragma unroll
    for (int nc = 0; nc < 4; ++nc) acc[mi][nc] = zero8();

  stage(0, 0);
  int cur = 0;
  for (int kt = 0; kt < Dm / 32; ++kt) {
    wait_async_copies();
    __syncthreads();
    if (kt + 1 < Dm / 32) stage(cur ^ 1, kt + 1);

    v16bf a[2];
#pragma unroll
    for (int mi = 0; mi < 2; ++mi)
      a[mi] = load_a_frag(&cs[cur][wid * 32 + mi * 16][0], 40);
#pragma unroll
    for (int nc = 0; nc < 4; ++nc) {
      v16bf bfrag = load_bt_frag(&wsm[cur][nc * 16][0], 40);
#pragma unroll
      for (int mi = 0; mi < 2; ++mi)
        acc[mi][nc] = wmma_bf16(a[mi], bfrag, acc[mi][nc]);
    }
    cur ^= 1;
  }

  const int nl = lane & 15;
  const int mh = (lane & 16) ? 8 : 0;
#pragma unroll
  for (int mi = 0; mi < 2; ++mi)
#pragma unroll
    for (int nc = 0; nc < 4; ++nc)
#pragma unroll
      for (int v = 0; v < 8; ++v) {
        int gm = m0 + wid * 32 + mi * 16 + mh + v;
        int n  = n0 + nc * 16 + nl;
        out[(size_t)gm * Dm + n] = acc[mi][nc][v] + bo[n];
      }
}

// ---------------- launch ----------------
extern "C" void kernel_launch(void* const* d_in, const int* in_sizes, int n_in,
                              void* d_out, int out_size, void* d_ws,
                              size_t ws_size, hipStream_t stream) {
  (void)in_sizes; (void)n_in; (void)out_size; (void)ws_size;
  const float* x  = (const float*)d_in[0];
  const float* Wq = (const float*)d_in[1];
  const float* bq = (const float*)d_in[2];
  const float* Wk = (const float*)d_in[3];
  const float* bk = (const float*)d_in[4];
  const float* Wv = (const float*)d_in[5];
  const float* bv = (const float*)d_in[6];
  const float* Wo = (const float*)d_in[7];
  const float* bo = (const float*)d_in[8];
  float* out = (float*)d_out;

  char* wsb = (char*)d_ws;
  const size_t MB = 1ull << 20;
  bf16* wh  = (bf16*)(wsb + 0 * MB);    // 4 x D*D bf16  (8 MB)
  bf16* qh  = (bf16*)(wsb + 8 * MB);    // [B,H,S,DK]    (16 MB)
  bf16* kh  = (bf16*)(wsb + 24 * MB);   // [B,H,S,DK]    (16 MB)
  bf16* vt  = (bf16*)(wsb + 40 * MB);   // [B,H,DK,S]    (16 MB)
  bf16* ctx = (bf16*)(wsb + 56 * MB);   // [B,S,D]       (16 MB)
  bf16* xh  = (bf16*)(wsb + 72 * MB);   // [B,S,D]       (16 MB)

  const int DD = Dm * Dm;
  cvt_f32_bf16_v4<<<1024, 256, 0, stream>>>(Wq, wh + 0 * (size_t)DD, DD / 4);
  cvt_f32_bf16_v4<<<1024, 256, 0, stream>>>(Wk, wh + 1 * (size_t)DD, DD / 4);
  cvt_f32_bf16_v4<<<1024, 256, 0, stream>>>(Wv, wh + 2 * (size_t)DD, DD / 4);
  cvt_f32_bf16_v4<<<1024, 256, 0, stream>>>(Wo, wh + 3 * (size_t)DD, DD / 4);
  cvt_f32_bf16_v4<<<2048, 256, 0, stream>>>(x, xh, (BATCH * S * Dm) / 4);

  qkv_gemm_kernel<<<dim3((BATCH * S) / 128, Dm / 64, 3), 128, 0, stream>>>(
      xh, wh, bq, bk, bv, qh, kh, vt);

  attn_kernel<<<dim3(S / 64, BATCH * H), 128, 0, stream>>>(qh, kh, vt, ctx);

  oproj_kernel<<<dim3((BATCH * S) / 128, Dm / 64), 128, 0, stream>>>(
      ctx, wh + 3 * (size_t)DD, bo, out);
}
